// SwinTransformerBlock_39341900431779
// MI455X (gfx1250) — matrix operand order repaired and checked by reading, not compiled, not run on hardware
//
#include <hip/hip_runtime.h>
#include <math.h>

// ---------------------------------------------------------------------------
// Types for CDNA5 WMMA (gfx1250, wave32)
// ---------------------------------------------------------------------------
typedef __attribute__((ext_vector_type(16))) __bf16 v16bf;
typedef __attribute__((ext_vector_type(8)))  float  v8f;
typedef __attribute__((ext_vector_type(4)))  int    v4i;

union FragAB {
    uint4 q[2];   // 32 bytes
    v16bf v;      // 16 x bf16
};

// Async global->LDS staging (ASYNCcnt path), guarded so the build never breaks.
#if defined(__HIP_DEVICE_COMPILE__) && __has_builtin(__builtin_amdgcn_global_load_async_to_lds_b128)
#define USE_ASYNC_LDS 1
typedef __attribute__((address_space(1))) v4i gas_v4i;
typedef __attribute__((address_space(3))) v4i las_v4i;
#else
#define USE_ASYNC_LDS 0
#endif

__device__ __forceinline__ void async_wait_all() {
#if USE_ASYNC_LDS
#if __has_builtin(__builtin_amdgcn_s_wait_asynccnt)
    __builtin_amdgcn_s_wait_asynccnt(0);
#else
    asm volatile("s_wait_asynccnt 0x0" ::: "memory");
#endif
#endif
}

__device__ __forceinline__ unsigned short f2bf(float f) {
    union { float f; unsigned int u; } c; c.f = f;
    unsigned int u = c.u;
    unsigned int r = u + 0x7FFFu + ((u >> 16) & 1u);
    return (unsigned short)(r >> 16);
}
__device__ __forceinline__ float bf2f(unsigned short h) {
    union { unsigned int u; float f; } c; c.u = ((unsigned int)h) << 16;
    return c.f;
}

// Problem constants
#define BATCH 32
#define HH 56
#define WW 56
#define CH 384
#define NWIN 2048        // 32 * 8 * 8
#define NTOK 49
#define MTOT 100352      // NWIN * NTOK
#define NHEAD 12
#define HD 32
#define DFF 1536

// windowed token m -> flat pixel index base (in units of CH floats)
__device__ __forceinline__ size_t map_token_to_pixel(int m) {
    int wi = m / NTOK, n = m % NTOK;
    int b  = wi >> 6;           // /64
    int r  = wi & 63;
    int hb = r >> 3, wb = r & 7;
    int hs = n / 7,  wsub = n % 7;
    int hh = hb * 7 + hs, ww = wb * 7 + wsub;
    return (((size_t)b * HH + hh) * WW + ww) * CH;
}

// ---------------------------------------------------------------------------
// Weight transpose + cast:  W[K][N] (f32) -> WT[N][K] (bf16)
// ---------------------------------------------------------------------------
__global__ void transpose_cast_kernel(const float* __restrict__ W,
                                      unsigned short* __restrict__ WT,
                                      int K, int N) {
    int i = blockIdx.x * 256 + threadIdx.x;
    if (i < N * K) {
        int n = i / K, k = i - n * K;
        WT[i] = f2bf(W[(size_t)k * N + n]);
    }
}

// ---------------------------------------------------------------------------
// LayerNorm (one wave per token).  MAP=true: gather input through the window
// permute (LN1 on x).  MAP=false: identity (LN2 on resid1).  Output bf16.
// ---------------------------------------------------------------------------
template <bool MAP>
__global__ __launch_bounds__(128)
void layernorm_kernel(const float* __restrict__ in,
                      const float* __restrict__ gamma,
                      const float* __restrict__ beta,
                      unsigned short* __restrict__ out) {
    int m    = blockIdx.x * 4 + (threadIdx.x >> 5);
    int lane = threadIdx.x & 31;
    size_t inBase = MAP ? map_token_to_pixel(m) : (size_t)m * CH;

    float v[12];
    float s = 0.f, s2 = 0.f;
#pragma unroll
    for (int i = 0; i < 12; ++i) {
        float t = in[inBase + lane + i * 32];
        v[i] = t; s += t; s2 += t * t;
    }
#pragma unroll
    for (int o = 16; o > 0; o >>= 1) {
        s  += __shfl_xor(s,  o, 32);
        s2 += __shfl_xor(s2, o, 32);
    }
    float mean = s * (1.f / CH);
    float var  = s2 * (1.f / CH) - mean * mean;
    float rstd = rsqrtf(var + 1e-5f);
#pragma unroll
    for (int i = 0; i < 12; ++i) {
        int c = lane + i * 32;
        out[(size_t)m * CH + c] = f2bf((v[i] - mean) * rstd * gamma[c] + beta[c]);
    }
}

// ---------------------------------------------------------------------------
// Tiled bf16 WMMA GEMM:  C[M x N] = A[M x K] * Bt[N x K]^T  (+ epilogue)
// BM=BN=128, BK=32.  256 threads = 8 waves in 2x4; each wave does 64x32
// (4x2 tiles of v_wmma_f32_16x16x32_bf16).  Double-buffered LDS staging via
// GLOBAL_LOAD_ASYNC_TO_LDS_B128 when available (ASYNCcnt), sync copy otherwise.
// ---------------------------------------------------------------------------
#define EPI_QKV  0
#define EPI_PROJ 1
#define EPI_MLP1 2
#define EPI_MLP2 3

template <int EPI>
__global__ __launch_bounds__(256)
void gemm_bf16_kernel(const unsigned short* __restrict__ A,
                      const unsigned short* __restrict__ Bt,
                      const float* __restrict__ bias,
                      const float* __restrict__ extra,   // PROJ: x ; MLP2: resid1
                      float* __restrict__ outF,          // PROJ: resid1 ; MLP2: d_out
                      unsigned short* __restrict__ outB0,// QKV: Q ; MLP1: H
                      unsigned short* __restrict__ outB1,// QKV: K
                      unsigned short* __restrict__ outB2,// QKV: V
                      int K, int N) {
    __shared__ __align__(16) unsigned short As[2][128][40];
    __shared__ __align__(16) unsigned short Bs[2][128][40];

    const int tid  = threadIdx.x;
    const int lane = tid & 31;
    const int wave = tid >> 5;
    const int m0 = blockIdx.y * 128;
    const int n0 = blockIdx.x * 128;
    const int wm = (wave >> 2) * 64;   // 0 / 64
    const int wn = (wave & 3) * 32;    // 0,32,64,96
    const int hl  = lane >> 4;         // half: 0/1
    const int l15 = lane & 15;

    // stage one 128x32 A tile and one 128x32 Bt tile into LDS buffer `buf`
    auto stage = [&](int k0, int buf) {
#pragma unroll
        for (int rep = 0; rep < 2; ++rep) {
            int idx = tid + rep * 256;           // 0..511
            int r = idx >> 2, c4 = idx & 3;      // row, 8-elem column group
            const unsigned short* ga = A  + (size_t)(m0 + r) * K + k0 + c4 * 8;
            const unsigned short* gb = Bt + (size_t)(n0 + r) * K + k0 + c4 * 8;
            unsigned short* la = &As[buf][r][c4 * 8];
            unsigned short* lb = &Bs[buf][r][c4 * 8];
#if USE_ASYNC_LDS
            __builtin_amdgcn_global_load_async_to_lds_b128(
                (gas_v4i*)ga, (las_v4i*)la, 0, 0);
            __builtin_amdgcn_global_load_async_to_lds_b128(
                (gas_v4i*)gb, (las_v4i*)lb, 0, 0);
#else
            *reinterpret_cast<uint4*>(la) = *reinterpret_cast<const uint4*>(ga);
            *reinterpret_cast<uint4*>(lb) = *reinterpret_cast<const uint4*>(gb);
#endif
        }
    };

    v8f acc[4][2];
#pragma unroll
    for (int mi = 0; mi < 4; ++mi)
#pragma unroll
        for (int ni = 0; ni < 2; ++ni)
#pragma unroll
            for (int e = 0; e < 8; ++e) acc[mi][ni][e] = 0.f;

    stage(0, 0);
    async_wait_all();
    __syncthreads();

    int buf = 0;
    for (int k0 = 0; k0 < K; k0 += 32) {
        // prefetch next K-tile into the other buffer while we compute
        if (k0 + 32 < K) stage(k0 + 32, buf ^ 1);

        FragAB a[4], b[2];
#pragma unroll
        for (int mi = 0; mi < 4; ++mi) {
            int row = wm + mi * 16 + l15;
            a[mi].q[0] = *reinterpret_cast<const uint4*>(&As[buf][row][hl * 8]);
            a[mi].q[1] = *reinterpret_cast<const uint4*>(&As[buf][row][16 + hl * 8]);
        }
#pragma unroll
        for (int ni = 0; ni < 2; ++ni) {
            int row = wn + ni * 16 + l15;
            b[ni].q[0] = *reinterpret_cast<const uint4*>(&Bs[buf][row][hl * 8]);
            b[ni].q[1] = *reinterpret_cast<const uint4*>(&Bs[buf][row][16 + hl * 8]);
        }
#pragma unroll
        for (int mi = 0; mi < 4; ++mi)
#pragma unroll
            for (int ni = 0; ni < 2; ++ni)
                acc[mi][ni] = __builtin_amdgcn_wmma_f32_16x16x32_bf16(
                    false, a[mi].v, false, b[ni].v, (short)0, acc[mi][ni],
                    false, false);

        async_wait_all();     // our async writes into buf^1 are done
        __syncthreads();      // everyone finished reading buf, sees buf^1 filled
        buf ^= 1;
    }

    // epilogue: lane (0-15): col=lane, rows e=0..7 ; lane (16-31): rows 8+e
#pragma unroll
    for (int mi = 0; mi < 4; ++mi) {
#pragma unroll
        for (int ni = 0; ni < 2; ++ni) {
#pragma unroll
            for (int e = 0; e < 8; ++e) {
                int gm = m0 + wm + mi * 16 + hl * 8 + e;
                int gn = n0 + wn + ni * 16 + l15;
                float val = acc[mi][ni][e] + bias[gn];
                if (EPI == EPI_QKV) {
                    int which = (gn >= 768) ? 2 : ((gn >= 384) ? 1 : 0);
                    int c = gn - which * 384;
                    if (which == 0) val *= 0.17677669529663687f;  // hd^-0.5
                    int wi = gm / NTOK, n = gm - wi * NTOK;
                    int h = c >> 5, d = c & 31;
                    size_t idx = (((size_t)wi * NHEAD + h) * NTOK + n) * HD + d;
                    unsigned short* dst = (which == 0) ? outB0
                                        : (which == 1) ? outB1 : outB2;
                    dst[idx] = f2bf(val);
                } else if (EPI == EPI_PROJ) {
                    size_t xidx = map_token_to_pixel(gm) + gn;
                    outF[(size_t)gm * CH + gn] = val + extra[xidx];
                } else if (EPI == EPI_MLP1) {
                    // exact GELU
                    val = 0.5f * val * (1.f + erff(val * 0.7071067811865476f));
                    outB0[(size_t)gm * DFF + gn] = f2bf(val);
                } else { // EPI_MLP2
                    val += extra[(size_t)gm * CH + gn];
                    size_t oidx = map_token_to_pixel(gm) + gn;
                    outF[oidx] = val;
                }
            }
        }
    }
}

// ---------------------------------------------------------------------------
// Windowed attention: one block (128 thr = 4 waves) per (window, head).
// Pads 49 tokens -> 64.  Scores + PV via WMMA, softmax in LDS.
// ---------------------------------------------------------------------------
__global__ __launch_bounds__(128)
void attn_kernel(const unsigned short* __restrict__ Qb,
                 const unsigned short* __restrict__ Kb,
                 const unsigned short* __restrict__ Vb,
                 unsigned short* __restrict__ AttnOut) {
    const int wh = blockIdx.x;            // window*NHEAD + head
    const int wi = wh / NHEAD, h = wh - wi * NHEAD;
    const size_t base = (size_t)wh * NTOK * HD;

    __shared__ __align__(16) unsigned short Qs[64][40];
    __shared__ __align__(16) unsigned short Ks[64][40];
    __shared__ __align__(16) unsigned short Vt[32][72];  // [d][token]
    __shared__ __align__(16) unsigned short Ps[64][72];
    __shared__ float Ss[64][68];

    const int tid = threadIdx.x;
    // stage Q, K (zero-pad rows 49..63)
    for (int e = tid; e < 64 * 32; e += 128) {
        int r = e >> 5, d = e & 31;
        unsigned short q = 0, k = 0;
        if (r < NTOK) {
            q = Qb[base + r * HD + d];
            k = Kb[base + r * HD + d];
        }
        Qs[r][d] = q;
        Ks[r][d] = k;
    }
    // stage V transposed (zero-pad tokens 49..63)
    for (int e = tid; e < 32 * 64; e += 128) {
        int d = e >> 6, r = e & 63;
        Vt[d][r] = (r < NTOK) ? Vb[base + r * HD + d] : (unsigned short)0;
    }
    __syncthreads();

    const int wave = tid >> 5, lane = tid & 31;
    const int hl = lane >> 4, l15 = lane & 15;
    const int m0 = wave * 16;             // this wave's row block

    // ---- scores S = Q * K^T  (4 j-tiles) ----
    FragAB aq;
    {
        int row = m0 + l15;
        aq.q[0] = *reinterpret_cast<const uint4*>(&Qs[row][hl * 8]);
        aq.q[1] = *reinterpret_cast<const uint4*>(&Qs[row][16 + hl * 8]);
    }
#pragma unroll
    for (int j = 0; j < 4; ++j) {
        FragAB bk;
        int row = j * 16 + l15;
        bk.q[0] = *reinterpret_cast<const uint4*>(&Ks[row][hl * 8]);
        bk.q[1] = *reinterpret_cast<const uint4*>(&Ks[row][16 + hl * 8]);
        v8f z;
#pragma unroll
        for (int e = 0; e < 8; ++e) z[e] = 0.f;
        v8f s = __builtin_amdgcn_wmma_f32_16x16x32_bf16(
            false, aq.v, false, bk.v, (short)0, z, false, false);
#pragma unroll
        for (int e = 0; e < 8; ++e)
            Ss[m0 + hl * 8 + e][j * 16 + l15] = s[e];
    }
    __syncthreads();

    // ---- masked softmax over cols [0,49); lanes 0..15 of each wave: 1 row ----
    if (lane < 16) {
        int row = m0 + lane;
        float mx = -1e30f;
        for (int c = 0; c < NTOK; ++c) mx = fmaxf(mx, Ss[row][c]);
        float sum = 0.f;
        for (int c = 0; c < NTOK; ++c) sum += expf(Ss[row][c] - mx);
        float inv = 1.f / sum;
        for (int c = 0; c < 64; ++c) {
            float p = (c < NTOK) ? expf(Ss[row][c] - mx) * inv : 0.f;
            Ps[row][c] = f2bf(p);
        }
    }
    __syncthreads();

    // ---- O = P * V  (K over 64 tokens = 2 chunks; 2 column tiles of d) ----
    v8f o[2];
#pragma unroll
    for (int ni = 0; ni < 2; ++ni)
#pragma unroll
        for (int e = 0; e < 8; ++e) o[ni][e] = 0.f;
#pragma unroll
    for (int kc = 0; kc < 2; ++kc) {
        FragAB ap;
        int row = m0 + l15;
        ap.q[0] = *reinterpret_cast<const uint4*>(&Ps[row][kc * 32 + hl * 8]);
        ap.q[1] = *reinterpret_cast<const uint4*>(&Ps[row][kc * 32 + 16 + hl * 8]);
#pragma unroll
        for (int ni = 0; ni < 2; ++ni) {
            FragAB bv;
            int drow = ni * 16 + l15;
            bv.q[0] = *reinterpret_cast<const uint4*>(&Vt[drow][kc * 32 + hl * 8]);
            bv.q[1] = *reinterpret_cast<const uint4*>(&Vt[drow][kc * 32 + 16 + hl * 8]);
            o[ni] = __builtin_amdgcn_wmma_f32_16x16x32_bf16(
                false, ap.v, false, bv.v, (short)0, o[ni], false, false);
        }
    }
    // ---- store: AttnOut[(wi*49+tok)*384 + h*32 + d] ----
#pragma unroll
    for (int ni = 0; ni < 2; ++ni) {
#pragma unroll
        for (int e = 0; e < 8; ++e) {
            int tok = m0 + hl * 8 + e;
            if (tok < NTOK) {
                int ch = h * HD + ni * 16 + l15;
                AttnOut[((size_t)wi * NTOK + tok) * CH + ch] = f2bf(o[ni][e]);
            }
        }
    }
}

// ---------------------------------------------------------------------------
// Host-side orchestration
// ---------------------------------------------------------------------------
extern "C" void kernel_launch(void* const* d_in, const int* in_sizes, int n_in,
                              void* d_out, int out_size, void* d_ws, size_t ws_size,
                              hipStream_t stream) {
    const float* x      = (const float*)d_in[0];
    const float* ln1_g  = (const float*)d_in[1];
    const float* ln1_b  = (const float*)d_in[2];
    const float* qkv_w  = (const float*)d_in[3];
    const float* qkv_b  = (const float*)d_in[4];
    const float* proj_w = (const float*)d_in[5];
    const float* proj_b = (const float*)d_in[6];
    const float* ln2_g  = (const float*)d_in[7];
    const float* ln2_b  = (const float*)d_in[8];
    const float* w1     = (const float*)d_in[9];
    const float* b1     = (const float*)d_in[10];
    const float* w2     = (const float*)d_in[11];
    const float* b2     = (const float*)d_in[12];
    float* out = (float*)d_out;

    char* ws = (char*)d_ws;
    size_t off = 0;
    auto alloc = [&](size_t bytes) -> void* {
        void* p = ws + off;
        off += (bytes + 255) & ~(size_t)255;
        return p;
    };

    unsigned short* qkvWT  = (unsigned short*)alloc((size_t)1152 * 384 * 2);
    unsigned short* projWT = (unsigned short*)alloc((size_t)384 * 384 * 2);
    unsigned short* w1T    = (unsigned short*)alloc((size_t)1536 * 384 * 2);
    unsigned short* w2T    = (unsigned short*)alloc((size_t)384 * 1536 * 2);
    const size_t actB = (size_t)MTOT * CH * 2;            // 77 MB (bf16)
    unsigned short* Xw   = (unsigned short*)alloc(actB);  // reused as Xn2
    unsigned short* Qp   = (unsigned short*)alloc(actB);  // Q,K,V,AttnOut contiguous:
    unsigned short* Kp   = (unsigned short*)alloc(actB);  //   reused as H (bf16, MTOT x 1536)
    unsigned short* Vp   = (unsigned short*)alloc(actB);
    unsigned short* AOut = (unsigned short*)alloc(actB);
    float*          res1 = (float*)alloc((size_t)MTOT * CH * 4);
    unsigned short* Hbuf = Qp;  // 4*actB = MTOT*1536 bf16
    unsigned short* Xn2  = Xw;

    // 1) weight transposes (f32 [K][N] -> bf16 [N][K])
    transpose_cast_kernel<<<(1152 * 384 + 255) / 256, 256, 0, stream>>>(qkv_w,  qkvWT,  384, 1152);
    transpose_cast_kernel<<<( 384 * 384 + 255) / 256, 256, 0, stream>>>(proj_w, projWT, 384,  384);
    transpose_cast_kernel<<<(1536 * 384 + 255) / 256, 256, 0, stream>>>(w1,     w1T,    384, 1536);
    transpose_cast_kernel<<<( 384 *1536 + 255) / 256, 256, 0, stream>>>(w2,     w2T,   1536,  384);

    // 2) LN1 + window partition -> Xw (bf16, windowed token order)
    layernorm_kernel<true><<<MTOT / 4, 128, 0, stream>>>(x, ln1_g, ln1_b, Xw);

    // 3) QKV GEMM (+bias, q-scale, scatter to [win][head][tok][d])
    gemm_bf16_kernel<EPI_QKV><<<dim3(1152 / 128, MTOT / 128), 256, 0, stream>>>(
        Xw, qkvWT, qkv_b, nullptr, nullptr, Qp, Kp, Vp, 384, 1152);

    // 4) windowed attention
    attn_kernel<<<NWIN * NHEAD, 128, 0, stream>>>(Qp, Kp, Vp, AOut);

    // 5) proj GEMM (+bias, +shortcut x via window map) -> resid1 (f32, windowed)
    gemm_bf16_kernel<EPI_PROJ><<<dim3(384 / 128, MTOT / 128), 256, 0, stream>>>(
        AOut, projWT, proj_b, x, res1, nullptr, nullptr, nullptr, 384, 384);

    // 6) LN2 -> Xn2 (bf16)
    layernorm_kernel<false><<<MTOT / 4, 128, 0, stream>>>(res1, ln2_g, ln2_b, Xn2);

    // 7) MLP1 GEMM (+bias, exact GELU) -> H (bf16)
    gemm_bf16_kernel<EPI_MLP1><<<dim3(1536 / 128, MTOT / 128), 256, 0, stream>>>(
        Xn2, w1T, b1, nullptr, nullptr, Hbuf, nullptr, nullptr, 384, 1536);

    // 8) MLP2 GEMM (+bias, +resid1, un-window) -> d_out (f32)
    gemm_bf16_kernel<EPI_MLP2><<<dim3(384 / 128, MTOT / 128), 256, 0, stream>>>(
        Hbuf, w2T, b2, res1, out, nullptr, nullptr, nullptr, 1536, 384);

    (void)in_sizes; (void)n_in; (void)out_size; (void)ws_size;
}
